// GlobalAggregator_65077344469222
// MI455X (gfx1250) — compile-verified, matrix operand order checked by reading, not run on hardware
//
#include <hip/hip_runtime.h>

typedef __attribute__((ext_vector_type(16))) _Float16 v16h;
typedef __attribute__((ext_vector_type(8)))  _Float16 v8h;
typedef __attribute__((ext_vector_type(8)))  float    v8f;

#define Bsz 128
#define Nn  128
#define Kn  12
#define Dd  256

// workspace byte offsets
#define W1B_OFF 0         // 16 tiles * 8 kblks * 32 lanes * 16 halves = 65536 f16 (131072 B)
#define W3B_OFF 131072    // 16 tiles * 16 kblks * 32 lanes * 16 halves = 131072 f16 (262144 B)
#define AGG_OFF 393216    // 16384 * 256 f16 = 8 MB

// ---------------------------------------------------------------------------
// Convert w1 (first 256 rows) and w3 to f16, pre-swizzled into the per-lane
// B-fragment layout of V_WMMA_F32_16X16X32_F16:
//   fragment index = ((tile*nK + j)*32 + lane)*16 + h
//   lane<16 : col = tile*16 + lane,   h<8 -> k = 32j+h,      h>=8 -> k = 32j+16+(h-8)
//   lane>=16: col = tile*16 + lane-16,h<8 -> k = 32j+8+h,    h>=8 -> k = 32j+24+(h-8)
// ---------------------------------------------------------------------------
__global__ void __launch_bounds__(256)
prep_weights(const float* __restrict__ w1, const float* __restrict__ w3,
             _Float16* __restrict__ w1B, _Float16* __restrict__ w3B) {
  int gid = blockIdx.x * blockDim.x + threadIdx.x;

  if (gid < 16 * 16 * 32 * 16) {          // w3B: 16 tiles x 16 kblks
    int h    = gid & 15;
    int lane = (gid >> 4) & 31;
    int j    = (gid >> 9) & 15;
    int t    = (gid >> 13) & 15;
    int half = (lane >> 4) & 1;
    int f    = j * 32 + ((h < 8) ? (half * 8 + h) : (16 + half * 8 + (h - 8)));
    int col  = t * 16 + (lane & 15);
    w3B[gid] = (_Float16)w3[f * Dd + col];
  }
  if (gid < 16 * 8 * 32 * 16) {           // w1B: 16 tiles x 8 kblks
    int h    = gid & 15;
    int lane = (gid >> 4) & 31;
    int j    = (gid >> 9) & 7;
    int t    = (gid >> 12) & 15;
    int half = (lane >> 4) & 1;
    int f    = j * 32 + ((h < 8) ? (half * 8 + h) : (16 + half * 8 + (h - 8)));
    int col  = t * 16 + (lane & 15);
    w1B[gid] = (_Float16)w1[f * Dd + col];
  }
}

// ---------------------------------------------------------------------------
// One block per (b,n).  LDS-resident neighbor tile is used twice (feature
// product and weighted aggregation) so HBM sees it exactly once.
// alpha1 = leaky_relu(feat @ w1) via v_wmma_f32_16x16x32_f16, K=12 rows padded
// to 16; the "+1" weight feature is folded into the C initializer (rank-1).
// ---------------------------------------------------------------------------
__global__ void __launch_bounds__(256)
alpha_agg_kernel(const float* __restrict__ nbr,
                 const float* __restrict__ nw,
                 const float* __restrict__ extra,
                 const float* __restrict__ w1,
                 const float* __restrict__ w2,
                 const _Float16* __restrict__ w1B,
                 _Float16* __restrict__ aggH) {
  __shared__ __align__(16) float    nbf[Kn * Dd];   // 12 KB, fp32 neighbor tile
  __shared__ float                  exv[Dd];
  __shared__ __align__(16) _Float16 feat[16 * Dd];  // 8 KB, padded f16 A tile
  __shared__ float nwS[16];
  __shared__ float alphaPre[16];
  __shared__ float alphaS[16];

  int bn   = blockIdx.x;
  int tid  = threadIdx.x;
  int lane = tid & 31;
  int wave = tid >> 5;
  int half = (lane >> 4) & 1;
  int r16  = lane & 15;

  const float* nbG = nbr + (size_t)bn * (Kn * Dd);
  for (int i = tid; i < Kn * Dd; i += 256) nbf[i] = nbG[i];
  exv[tid] = extra[(size_t)bn * Dd + tid];
  if (tid < 16) {
    nwS[tid]      = (tid < Kn) ? nw[(size_t)bn * Kn + tid] : 0.f;
    alphaPre[tid] = 0.f;
  }
  __syncthreads();

  // feat[r][f] = extra[f] * neighbor[r][f] in f16, rows 12..15 zero-padded
  for (int r = 0; r < 16; ++r) {
    float v = (r < Kn) ? exv[tid] * nbf[r * Dd + tid] : 0.f;
    feat[r * Dd + tid] = (_Float16)v;
  }
  __syncthreads();

  float accR[8];
  #pragma unroll
  for (int v = 0; v < 8; ++v) accR[v] = 0.f;

  for (int tt = 0; tt < 2; ++tt) {
    int t = wave * 2 + tt;          // column tile 0..15
    int d = t * 16 + r16;           // this lane's output column
    float w1last = w1[256 * Dd + d];
    float w2d    = w2[d];

    v8f c;
    #pragma unroll
    for (int v = 0; v < 8; ++v) c[v] = nwS[v + half * 8] * w1last;  // rank-1 init

    #pragma unroll
    for (int j = 0; j < 8; ++j) {
      v8h lo = *(const v8h*)&feat[r16 * Dd + j * 32 + half * 8];
      v8h hi = *(const v8h*)&feat[r16 * Dd + j * 32 + 16 + half * 8];
      v16h a;
      #pragma unroll
      for (int i = 0; i < 8; ++i) { a[i] = lo[i]; a[i + 8] = hi[i]; }
      v16h b = *(const v16h*)&w1B[((t * 8 + j) * 32 + lane) * 16];
      c = __builtin_amdgcn_wmma_f32_16x16x32_f16(false, a, false, b,
                                                 (short)0, c, false, false);
    }
    #pragma unroll
    for (int v = 0; v < 8; ++v) {   // fused leaky_relu + (.) @ w2 epilogue
      float x  = c[v];
      float lr = (x > 0.f) ? x : 0.2f * x;
      accR[v] += lr * w2d;
    }
  }
  #pragma unroll
  for (int v = 0; v < 8; ++v) atomicAdd(&alphaPre[v + half * 8], accR[v]);
  __syncthreads();

  if (tid == 0) {                    // softmax over K=12
    float mx = alphaPre[0];
    for (int k = 1; k < Kn; ++k) mx = fmaxf(mx, alphaPre[k]);
    float s = 0.f;
    for (int k = 0; k < Kn; ++k) { float e = __expf(alphaPre[k] - mx); alphaS[k] = e; s += e; }
    float inv = 1.f / s;
    for (int k = 0; k < Kn; ++k) alphaS[k] *= inv;
  }
  __syncthreads();

  float aggd = 0.f;                  // reuse fp32 neighbor tile from LDS
  for (int k = 0; k < Kn; ++k) aggd += alphaS[k] * nbf[k * Dd + tid];
  aggH[(size_t)bn * Dd + tid] = (_Float16)aggd;
}

// ---------------------------------------------------------------------------
// out = relu(concat(self, agg) @ w3).  16-row M tiles over 16384 (b,n) rows.
// ---------------------------------------------------------------------------
__global__ void __launch_bounds__(256)
out_gemm_kernel(const float* __restrict__ self_v,
                const _Float16* __restrict__ aggH,
                const _Float16* __restrict__ w3B,
                float* __restrict__ out) {
  __shared__ __align__(16) _Float16 cat[16 * 512];  // 16 KB

  int mt   = blockIdx.x;
  int tid  = threadIdx.x;
  int lane = tid & 31;
  int wave = tid >> 5;
  int half = (lane >> 4) & 1;
  int r16  = lane & 15;
  int m0   = mt * 16;

  for (int r = 0; r < 16; ++r) {
    size_t m = (size_t)(m0 + r);
    cat[r * 512 + tid]       = (_Float16)self_v[m * Dd + tid];
    cat[r * 512 + 256 + tid] = aggH[m * Dd + tid];
  }
  __syncthreads();

  for (int tt = 0; tt < 2; ++tt) {
    int t = wave * 2 + tt;
    v8f c = {};
    #pragma unroll
    for (int j = 0; j < 16; ++j) {
      v8h lo = *(const v8h*)&cat[r16 * 512 + j * 32 + half * 8];
      v8h hi = *(const v8h*)&cat[r16 * 512 + j * 32 + 16 + half * 8];
      v16h a;
      #pragma unroll
      for (int i = 0; i < 8; ++i) { a[i] = lo[i]; a[i + 8] = hi[i]; }
      v16h b = *(const v16h*)&w3B[((t * 16 + j) * 32 + lane) * 16];
      c = __builtin_amdgcn_wmma_f32_16x16x32_f16(false, a, false, b,
                                                 (short)0, c, false, false);
    }
    int d = t * 16 + r16;
    #pragma unroll
    for (int v = 0; v < 8; ++v) {
      int m   = m0 + v + half * 8;
      float x = c[v];
      out[(size_t)m * Dd + d] = (x > 0.f) ? x : 0.f;   // fused ReLU
    }
  }
}

// ---------------------------------------------------------------------------
extern "C" void kernel_launch(void* const* d_in, const int* in_sizes, int n_in,
                              void* d_out, int out_size, void* d_ws, size_t ws_size,
                              hipStream_t stream) {
  const float* self_v = (const float*)d_in[0];
  const float* nbr    = (const float*)d_in[1];
  // d_in[2] masks: all-ones, unused by the reference math
  const float* nw     = (const float*)d_in[3];
  const float* extra  = (const float*)d_in[4];
  const float* w1     = (const float*)d_in[5];
  const float* w2     = (const float*)d_in[6];
  const float* w3     = (const float*)d_in[7];
  float* out = (float*)d_out;

  char* ws = (char*)d_ws;
  _Float16* w1B  = (_Float16*)(ws + W1B_OFF);
  _Float16* w3B  = (_Float16*)(ws + W3B_OFF);
  _Float16* aggH = (_Float16*)(ws + AGG_OFF);

  prep_weights<<<(16 * 16 * 32 * 16) / 256, 256, 0, stream>>>(w1, w3, w1B, w3B);
  alpha_agg_kernel<<<Bsz * Nn, 256, 0, stream>>>(nbr, nw, extra, w1, w2, w1B, aggH);
  out_gemm_kernel<<<(Bsz * Nn) / 16, 256, 0, stream>>>(self_v, aggH, w3B, out);
}
